// MLA_83391085019236
// MI455X (gfx1250) — matrix operand order repaired and checked by reading, not compiled, not run on hardware
//
#include <hip/hip_runtime.h>
#include <hip/hip_bf16.h>

namespace {

constexpr int B = 2, S = 2048, DIM = 2048, H = 16;
constexpr int NOPE = 128, ROPE = 64, VD = 128, QKD = NOPE + ROPE;
constexpr int QLORA = 768, KVLORA = 512;
constexpr float EPS = 1e-6f;
constexpr float SCALE = 0.07216878364870322f;  // QKD^-0.5 = 192^-0.5

typedef __attribute__((ext_vector_type(16))) __bf16 v16bf;
typedef __attribute__((ext_vector_type(8)))  float  v8f;

__device__ __forceinline__ v8f wmma_bf16(v16bf a, v16bf b, v8f c) {
  // D = A(16x32 bf16) * B(32x16 bf16) + C(16x16 f32)
  return __builtin_amdgcn_wmma_f32_16x16x32_bf16(false, a, false, b, (short)0, c,
                                                 false, false);
}

// ---------------------------------------------------------------------------
// GEMM: C[M,N] = A[M,K] @ W[K,N]; fp32 in/out, bf16 WMMA compute.
// 256 threads = 8 waves arranged 2(M) x 4(N); each wave owns a
// (WM*16) x (WN*16) register tile -> block tile (2*WM*16) x (4*WN*16).
// A fragments are reused across WN wmmas, B fragments across WM wmmas.
// Fragment layouts as loaded here (A and B share contiguous k, not the ISA table order):
//   A (16-bit, 16x32): lane m = L%16, half = L/16;
//     vgpr j holds K = half*16 + 2j (+1)   [pairs contiguous]
//   B (16-bit, 32x16): lane n = L%16, half = L/16; vgpr j holds K = half*16+2j (+1)
//   D (f32 16x16): lane = n + 16*(m/8); vgpr r -> m = r + 8*(lane/16)
// ---------------------------------------------------------------------------
template <int WM, int WN>
__global__ void __launch_bounds__(256)
gemm_wmma(const float* __restrict__ A, const float* __restrict__ W,
          float* __restrict__ C, int M, int N, int K) {
  const int lane = threadIdx.x & 31;
  const int wave = threadIdx.x >> 5;
  const int half = lane >> 4;
  const int l16  = lane & 15;
  const int tileM = (blockIdx.y * 2 + (wave >> 2)) * (WM * 16);
  const int tileN = (blockIdx.x * 4 + (wave & 3)) * (WN * 16);

  const float2* __restrict__ Arow[WM];
#pragma unroll
  for (int mi = 0; mi < WM; ++mi)
    Arow[mi] = (const float2*)(A + (size_t)(tileM + mi * 16 + l16) * K);

  v8f acc[WM][WN] = {};
  for (int kk = 0; kk < K; kk += 32) {
    v16bf a[WM], bf[WN];
#pragma unroll
    for (int mi = 0; mi < WM; ++mi)
#pragma unroll
      for (int j = 0; j < 8; ++j) {
        const int ka = kk + half * 16 + 2 * j;
        const float2 t = Arow[mi][ka >> 1];
        a[mi][2 * j]     = (__bf16)t.x;
        a[mi][2 * j + 1] = (__bf16)t.y;
      }
#pragma unroll
    for (int ni = 0; ni < WN; ++ni)
#pragma unroll
      for (int j = 0; j < 8; ++j) {
        const int kb = kk + half * 16 + 2 * j;
        const float* wp = W + (size_t)kb * N + tileN + ni * 16 + l16;
        bf[ni][2 * j]     = (__bf16)wp[0];
        bf[ni][2 * j + 1] = (__bf16)wp[N];
      }
#pragma unroll
    for (int mi = 0; mi < WM; ++mi)
#pragma unroll
      for (int ni = 0; ni < WN; ++ni)
        acc[mi][ni] = wmma_bf16(a[mi], bf[ni], acc[mi][ni]);
  }
#pragma unroll
  for (int mi = 0; mi < WM; ++mi)
#pragma unroll
    for (int ni = 0; ni < WN; ++ni)
#pragma unroll
      for (int r = 0; r < 8; ++r)
        C[(size_t)(tileM + mi * 16 + r + 8 * half) * N + tileN + ni * 16 + l16] =
            acc[mi][ni][r];
}

// ---------------------------------------------------------------------------
// Row-wise RMSNorm in place: one 256-thread block per row.
// ---------------------------------------------------------------------------
__global__ void __launch_bounds__(256)
rmsnorm_rows(float* __restrict__ x, const float* __restrict__ g, int n) {
  __shared__ float red[256];
  float* p = x + (size_t)blockIdx.x * n;
  float ss = 0.f;
  for (int i = threadIdx.x; i < n; i += 256) { const float t = p[i]; ss += t * t; }
  red[threadIdx.x] = ss;
  __syncthreads();
  for (int s = 128; s > 0; s >>= 1) {
    if ((int)threadIdx.x < s) red[threadIdx.x] += red[threadIdx.x + s];
    __syncthreads();
  }
  const float inv = rsqrtf(red[0] / (float)n + EPS);
  for (int i = threadIdx.x; i < n; i += 256) p[i] = p[i] * inv * g[i];
}

// ---------------------------------------------------------------------------
// RoPE in place on a 64-wide block at x + row*rowStride + peOffset.
// Lane d rotates pair (d, d+32) with angle pos * theta^(-d/32).
// ---------------------------------------------------------------------------
__global__ void __launch_bounds__(32)
rope_inplace(float* __restrict__ x, int rowStride, int peOffset, int headsPerPos) {
  const size_t row = blockIdx.x;
  const int d = threadIdx.x;  // 0..31
  const int pos = (int)((row / (size_t)headsPerPos) % (size_t)S);
  float* p = x + row * (size_t)rowStride + peOffset;
  const float inv_freq = __powf(10000.0f, -(float)d * (1.0f / 32.0f));
  float sv, cv;
  __sincosf((float)pos * inv_freq, &sv, &cv);
  const float x1 = p[d], x2 = p[d + 32];
  p[d]      = x1 * cv - x2 * sv;
  p[d + 32] = x1 * sv + x2 * cv;
}

// ---------------------------------------------------------------------------
// Flash-style causal attention. One wave per (b, h, 32-q-row tile): two 16-row
// score subtiles share every K/Kpe/V fragment (halves K/V re-read traffic).
// Scores never touch global memory; P goes D-layout -> A-layout via 2KB LDS.
// Softmax denominator is kept as a per-lane partial (corr is uniform across the
// 16-lane row group) and reduced once in the epilogue -> only the max needs a
// per-tile cross-lane butterfly. Next key tile is prefetched with
// global_prefetch_b8 while the current tile's WMMAs run.
// ---------------------------------------------------------------------------
__global__ void __launch_bounds__(32)
mla_attn(const float* __restrict__ q,    // [B,S,H,QKD] rope-applied
         const float* __restrict__ kn,   // [B,S,H,NOPE]
         const float* __restrict__ kp,   // [B,S,ROPE] rope-applied
         const float* __restrict__ v,    // [B,S,H,VD]
         float* __restrict__ o) {        // [B,S,H*VD]
  __shared__ __bf16 Plds[2][16 * 32];
  const int lane = threadIdx.x;
  const int half = lane >> 4, l16 = lane & 15;
  const int qt = blockIdx.x;  // 32-row q tile
  const int h = blockIdx.y, b = blockIdx.z;

  // Resident Q fragments (A layout), pre-scaled by QKD^-0.5.
  v16bf qa[2][4], qp[2][2];
#pragma unroll
  for (int u = 0; u < 2; ++u) {
    const float2* qrow =
        (const float2*)(q + (((size_t)b * S + qt * 32 + u * 16 + l16) * H + h) * QKD);
#pragma unroll
    for (int f = 0; f < 4; ++f)
#pragma unroll
      for (int j = 0; j < 8; ++j) {
        const int d = f * 32 + half * 16 + 2 * j;
        const float2 t = qrow[d >> 1];
        qa[u][f][2 * j]     = (__bf16)(t.x * SCALE);
        qa[u][f][2 * j + 1] = (__bf16)(t.y * SCALE);
      }
#pragma unroll
    for (int f = 0; f < 2; ++f)
#pragma unroll
      for (int j = 0; j < 8; ++j) {
        const int d = NOPE + f * 32 + half * 16 + 2 * j;
        const float2 t = qrow[d >> 1];
        qp[u][f][2 * j]     = (__bf16)(t.x * SCALE);
        qp[u][f][2 * j + 1] = (__bf16)(t.y * SCALE);
      }
  }

  float rowmax[2][8], psum[2][8];  // psum: per-lane partial of the denominator
  v8f acc[2][8] = {};
#pragma unroll
  for (int u = 0; u < 2; ++u)
#pragma unroll
    for (int r = 0; r < 8; ++r) { rowmax[u][r] = -1e30f; psum[u][r] = 0.f; }

  const int ntile = qt + 1;  // 32-key tiles covering keys <= qt*32+31
  for (int jt = 0; jt < ntile; ++jt) {
    const int k0 = jt * 32;
    // Prefetch next key tile (one byte per lane pulls whole cache lines).
    if (jt + 1 < ntile) {
      const int kn0 = k0 + 32;
      const char* pk = (const char*)(kn + (((size_t)b * S + kn0 + lane) * H + h) * NOPE);
      const char* pv = (const char*)(v + (((size_t)b * S + kn0 + lane) * H + h) * VD);
      const char* pp = (const char*)(kp + ((size_t)b * S + kn0 + lane) * ROPE);
#pragma unroll
      for (int c = 0; c < 4; ++c) __builtin_prefetch(pk + c * 128, 0, 0);
#pragma unroll
      for (int c = 0; c < 4; ++c) __builtin_prefetch(pv + c * 128, 0, 0);
#pragma unroll
      for (int c = 0; c < 2; ++c) __builtin_prefetch(pp + c * 128, 0, 0);
    }
    v8f s[2][2] = {};
    // NOPE contribution: contract over d = 0..127; K fragments shared by both u.
#pragma unroll
    for (int f = 0; f < 4; ++f) {
      v16bf b0, b1;
#pragma unroll
      for (int j = 0; j < 8; ++j) {
        const int d = f * 32 + half * 16 + 2 * j;
        const float2 t0 = *(const float2*)(kn +
            (((size_t)b * S + k0 + l16) * H + h) * NOPE + d);
        const float2 t1 = *(const float2*)(kn +
            (((size_t)b * S + k0 + 16 + l16) * H + h) * NOPE + d);
        b0[2 * j] = (__bf16)t0.x; b0[2 * j + 1] = (__bf16)t0.y;
        b1[2 * j] = (__bf16)t1.x; b1[2 * j + 1] = (__bf16)t1.y;
      }
#pragma unroll
      for (int u = 0; u < 2; ++u) {
        s[u][0] = wmma_bf16(qa[u][f], b0, s[u][0]);
        s[u][1] = wmma_bf16(qa[u][f], b1, s[u][1]);
      }
    }
    // ROPE contribution: k_pe shared across heads.
#pragma unroll
    for (int f = 0; f < 2; ++f) {
      v16bf b0, b1;
#pragma unroll
      for (int j = 0; j < 8; ++j) {
        const int d = f * 32 + half * 16 + 2 * j;
        const float2 t0 = *(const float2*)(kp + ((size_t)b * S + k0 + l16) * ROPE + d);
        const float2 t1 = *(const float2*)(kp + ((size_t)b * S + k0 + 16 + l16) * ROPE + d);
        b0[2 * j] = (__bf16)t0.x; b0[2 * j + 1] = (__bf16)t0.y;
        b1[2 * j] = (__bf16)t1.x; b1[2 * j + 1] = (__bf16)t1.y;
      }
#pragma unroll
      for (int u = 0; u < 2; ++u) {
        s[u][0] = wmma_bf16(qp[u][f], b0, s[u][0]);
        s[u][1] = wmma_bf16(qp[u][f], b1, s[u][1]);
      }
    }
    // Causal mask: only the diagonal key tile needs it.
    if (jt == ntile - 1) {
#pragma unroll
      for (int u = 0; u < 2; ++u)
#pragma unroll
        for (int r = 0; r < 8; ++r) {
          const int m = qt * 32 + u * 16 + r + 8 * half;
          if (k0 + l16 > m)      s[u][0][r] = -1e30f;
          if (k0 + 16 + l16 > m) s[u][1][r] = -1e30f;
        }
    }
    // Online softmax per q-row (rows live in 16-lane halves of the D layout).
#pragma unroll
    for (int u = 0; u < 2; ++u)
#pragma unroll
      for (int r = 0; r < 8; ++r) {
        float mx = fmaxf(s[u][0][r], s[u][1][r]);
#pragma unroll
        for (int off = 1; off < 16; off <<= 1) mx = fmaxf(mx, __shfl_xor(mx, off, 16));
        const float nm = fmaxf(rowmax[u][r], mx);
        const float corr = __expf(rowmax[u][r] - nm);  // uniform across the row group
        rowmax[u][r] = nm;
        const float p0 = __expf(s[u][0][r] - nm);
        const float p1 = __expf(s[u][1][r] - nm);
        psum[u][r] = psum[u][r] * corr + (p0 + p1);    // per-lane partial, no reduce
        Plds[u][(r + 8 * half) * 32 + l16]      = (__bf16)p0;
        Plds[u][(r + 8 * half) * 32 + 16 + l16] = (__bf16)p1;
#pragma unroll
        for (int t = 0; t < 8; ++t) acc[u][t][r] *= corr;
      }
    __syncthreads();  // single-wave WG: acts as an LDS fence (dscnt wait)
    v16bf pa[2];
#pragma unroll
    for (int u = 0; u < 2; ++u)
#pragma unroll
      for (int j = 0; j < 8; ++j) {
        const int ka = half * 16 + 2 * j;
        pa[u][2 * j]     = Plds[u][l16 * 32 + ka];
        pa[u][2 * j + 1] = Plds[u][l16 * 32 + ka + 1];
      }
    // O += P(16x32) @ V(32xVD); V fragments shared by both q subtiles.
#pragma unroll
    for (int t = 0; t < 8; ++t) {
      v16bf vb;
#pragma unroll
      for (int j = 0; j < 8; ++j) {
        const int kl = half * 16 + 2 * j;
        vb[2 * j]     = (__bf16)v[(((size_t)b * S + k0 + kl) * H + h) * VD + t * 16 + l16];
        vb[2 * j + 1] = (__bf16)v[(((size_t)b * S + k0 + kl + 1) * H + h) * VD + t * 16 + l16];
      }
#pragma unroll
      for (int u = 0; u < 2; ++u) acc[u][t] = wmma_bf16(pa[u], vb, acc[u][t]);
    }
    __syncthreads();
  }
  // Epilogue: reduce the denominator once, normalize, write [B,S,H*VD].
#pragma unroll
  for (int u = 0; u < 2; ++u)
#pragma unroll
    for (int r = 0; r < 8; ++r) {
      float ps = psum[u][r];
#pragma unroll
      for (int off = 1; off < 16; off <<= 1) ps += __shfl_xor(ps, off, 16);
      const float inv = 1.0f / ps;
      const int m = qt * 32 + u * 16 + r + 8 * half;
#pragma unroll
      for (int t = 0; t < 8; ++t)
        o[((size_t)b * S + m) * (size_t)(H * VD) + h * VD + t * 16 + l16] =
            acc[u][t][r] * inv;
    }
}

}  // namespace

extern "C" void kernel_launch(void* const* d_in, const int* in_sizes, int n_in,
                              void* d_out, int out_size, void* d_ws, size_t ws_size,
                              hipStream_t stream) {
  (void)in_sizes; (void)n_in; (void)out_size; (void)ws_size;
  const float* x     = (const float*)d_in[0];
  const float* w_dq  = (const float*)d_in[1];
  const float* g_q   = (const float*)d_in[2];
  const float* w_uq  = (const float*)d_in[3];
  const float* w_dkv = (const float*)d_in[4];
  const float* w_kr  = (const float*)d_in[5];
  const float* g_kv  = (const float*)d_in[6];
  const float* w_uk  = (const float*)d_in[7];
  const float* w_uv  = (const float*)d_in[8];
  const float* w_o   = (const float*)d_in[9];
  float* out = (float*)d_out;

  const int M = B * S;  // 4096 token rows
  float* ws   = (float*)d_ws;
  float* cq   = ws;                           // M * QLORA
  float* ckv  = cq   + (size_t)M * QLORA;     // M * KVLORA
  float* kpe  = ckv  + (size_t)M * KVLORA;    // M * ROPE
  float* qbuf = kpe  + (size_t)M * ROPE;      // M * H * QKD
  float* kn   = qbuf + (size_t)M * H * QKD;   // M * H * NOPE
  float* vbuf = kn   + (size_t)M * H * NOPE;  // M * H * VD
  float* aout = vbuf + (size_t)M * H * VD;    // M * H * VD

  const dim3 blk(256);
  // Big GEMMs: 128x256 block tile (wave tile 64x64, 4x4 wmma accumulators).
  auto grid44 = [&](int n) { return dim3(n / 256, M / 128); };

  // Down-projections + rope key projection (weights L2-resident).
  gemm_wmma<4, 4><<<grid44(QLORA), blk, 0, stream>>>(x, w_dq, cq, M, QLORA, DIM);
  gemm_wmma<4, 4><<<grid44(KVLORA), blk, 0, stream>>>(x, w_dkv, ckv, M, KVLORA, DIM);
  // N=64: 64x64 block tile (wave tile 32x16).
  gemm_wmma<2, 1><<<dim3(1, M / 64), blk, 0, stream>>>(x, w_kr, kpe, M, ROPE, DIM);

  // RMSNorm latents in place.
  rmsnorm_rows<<<M, blk, 0, stream>>>(cq,  g_q,  QLORA);
  rmsnorm_rows<<<M, blk, 0, stream>>>(ckv, g_kv, KVLORA);

  // Up-projections.
  gemm_wmma<4, 4><<<grid44(H * QKD), blk, 0, stream>>>(cq, w_uq, qbuf, M, H * QKD, QLORA);
  gemm_wmma<4, 4><<<grid44(H * NOPE), blk, 0, stream>>>(ckv, w_uk, kn, M, H * NOPE, KVLORA);
  gemm_wmma<4, 4><<<grid44(H * VD), blk, 0, stream>>>(ckv, w_uv, vbuf, M, H * VD, KVLORA);

  // RoPE on q_pe (per head) and k_pe (shared).
  rope_inplace<<<M * H, 32, 0, stream>>>(qbuf, QKD, NOPE, H);
  rope_inplace<<<M,     32, 0, stream>>>(kpe, ROPE, 0, 1);

  // Fused causal attention: one wave per (b, h, 32-row q tile).
  mla_attn<<<dim3(S / 32, H, B), 32, 0, stream>>>(qbuf, kn, kpe, vbuf, aout);

  // Output projection.
  gemm_wmma<4, 4><<<grid44(DIM), blk, 0, stream>>>(aout, w_o, out, M, DIM, H * VD);
}
